// Set2SetMatching_11656541241539
// MI455X (gfx1250) — matrix operand order repaired and checked by reading, not compiled, not run on hardware
//
#include <hip/hip_runtime.h>
#include <hip/hip_bf16.h>
#include <math.h>

typedef __attribute__((ext_vector_type(16))) _Float16 v16h;
typedef __attribute__((ext_vector_type(8)))  float    v8f;

#define NCLASS 100
#define NSUP   5
#define NQ     64
#define NH     10         // real heads
#define NHP    16         // padded heads (WMMA tile dim)
#define D      512
#define QTOT   (NCLASS*NQ)   // 6400 queries
#define ROWS   (QTOT*NH)     // 64000 normalized query rows
#define EPSN   1e-8f
#define BSTRIDE 528          // LDS f16 row stride (512 + 16 pad, keeps 32B alignment)

// ---------------------------------------------------------------------------
// Kernel 1: prototypes = mean over 5 supports, L2-normalize, emit padded f16.
// grid = NCLASS*16 blocks (one per (class, padded-head)), 256 threads.
// ---------------------------------------------------------------------------
__global__ void proto_kernel(const float* __restrict__ zs,
                             _Float16* __restrict__ protoH) {
    int tile = blockIdx.x;
    int c  = tile >> 4;
    int hp = tile & 15;
    int t  = threadIdx.x;          // 256 threads, 2 elements each
    __shared__ float red[256];

    int k0 = t * 2;
    float v0 = 0.f, v1 = 0.f;
    if (hp < NH) {
        for (int s = 0; s < NSUP; ++s) {
            const float* row = zs + ((size_t)((c * NSUP + s) * NH + hp)) * D;
            v0 += row[k0];
            v1 += row[k0 + 1];
        }
        v0 *= 0.2f;
        v1 *= 0.2f;
    }
    red[t] = v0 * v0 + v1 * v1;
    __syncthreads();
    for (int off = 128; off > 0; off >>= 1) {
        if (t < off) red[t] += red[t + off];
        __syncthreads();
    }
    float scale = 1.0f / fmaxf(sqrtf(red[0]), EPSN);
    _Float16* dst = protoH + ((size_t)(c * NHP + hp)) * D;
    dst[k0]     = (_Float16)(v0 * scale);   // hp>=10 rows write exact zeros
    dst[k0 + 1] = (_Float16)(v1 * scale);
}

// ---------------------------------------------------------------------------
// Kernel 2: per-row inverse L2 norm of zq. One wave32 per row.
// grid = ROWS/8 blocks, 256 threads (8 waves).
// ---------------------------------------------------------------------------
__global__ void invnorm_kernel(const float* __restrict__ zq,
                               float* __restrict__ invn) {
    int wave = threadIdx.x >> 5;
    int lane = threadIdx.x & 31;
    int row  = blockIdx.x * 8 + wave;
    const float* p = zq + (size_t)row * D;
    float s = 0.f;
    for (int j = 0; j < D; j += 32) {
        float x = p[j + lane];
        s += x * x;
    }
    for (int m = 16; m > 0; m >>= 1) s += __shfl_xor(s, m, 32);
    if (lane == 0) invn[row] = 1.0f / fmaxf(sqrtf(s), EPSN);
}

// ---------------------------------------------------------------------------
// B-tile staging helpers: 16 rows x 512 f16 = 1024 float4, 4 per thread.
// Fully unrolled so no exec-mask loop is generated.
// ---------------------------------------------------------------------------
__device__ __forceinline__ void tile_global_load(const float4* __restrict__ src,
                                                 int t, float4 st[4]) {
#pragma unroll
    for (int j = 0; j < 4; ++j) st[j] = src[t + 256 * j];
}

__device__ __forceinline__ void tile_lds_store(_Float16* dst, int t,
                                               const float4 st[4]) {
#pragma unroll
    for (int j = 0; j < 4; ++j) {
        int i   = t + 256 * j;
        int row = i >> 6;        // 64 float4 per 512-f16 row
        int col = i & 63;
        *(float4*)(&dst[row * BSTRIDE + col * 8]) = st[j];
    }
}

// ---------------------------------------------------------------------------
// Kernel 3: fused WMMA GEMM + max-over-proto-heads + sum-over-query-heads.
// One wave per query; 8 queries per workgroup share the class B tile in LDS.
// Each WMMA 16x16 output tile == one (query, class) pair's full head-x-head
// similarity block (head dims padded 10 -> 16 with zeros). Class tiles are
// double-buffered: global loads for class c+1 are issued before the WMMA
// block for class c, so L2 latency hides behind 16 WMMAs. (Deeper register
// pipelining of B fragments was tried and rejected: it pushed allocation past
// the register budget and spilled the prefetch staging to scratch.)
// ---------------------------------------------------------------------------
__global__ void __launch_bounds__(256)
gemm_kernel(const float* __restrict__ zq,
            const float* __restrict__ invn,
            const _Float16* __restrict__ protoH,
            float* __restrict__ scores) {
    __shared__ _Float16 ldsB[2][16 * BSTRIDE];   // 2 x 16.5 KB class tiles

    int t    = threadIdx.x;
    int wave = t >> 5;
    int lane = t & 31;
    int q    = blockIdx.x * 8 + wave;
    int l16  = lane & 15;         // A: row hq ; B: col hp  (per ISA layout)
    int half = lane >> 4;

    // ---- load this wave's A fragments once (kept in 128 VGPRs) ----
    // A 16x32 fragment, contiguous k order (same as B): lane L holds row L%16, K = (L/16)*16+{0..15}
    bool  valid = (l16 < NH);
    int   hqc   = valid ? l16 : 0;
    const float* rowp = zq + ((size_t)(q * NH + hqc)) * D;
    float sc = valid ? invn[q * NH + hqc] : 0.0f;   // padded rows -> zeros

    v16h afrag[16];
#pragma unroll
    for (int kc = 0; kc < 16; ++kc) {
        int k0 = kc * 32 + half * 16;
        v16h a;
#pragma unroll
        for (int i = 0; i < 8; ++i) a[i]     = (_Float16)(rowp[k0 + i]      * sc);
#pragma unroll
        for (int i = 0; i < 8; ++i) a[i + 8] = (_Float16)(rowp[k0 + 8 + i] * sc);
        afrag[kc] = a;
    }

    // ---- prologue: stage class 0 tile ----
    float4 st[4];
    tile_global_load((const float4*)protoH, t, st);
    tile_lds_store(&ldsB[0][0], t, st);
    __syncthreads();

    const int brow = l16 * BSTRIDE + half * 16;   // per-lane B base offset

    for (int c = 0; c < NCLASS; ++c) {
        const _Float16* buf = &ldsB[c & 1][0];

        // issue next tile's global loads before the math (latency hiding)
        if (c + 1 < NCLASS)
            tile_global_load((const float4*)(protoH + (size_t)(c + 1) * NHP * D),
                             t, st);

        // two independent accumulator chains over even/odd k-fragments
        v8f acc0, acc1;
#pragma unroll
        for (int r = 0; r < 8; ++r) { acc0[r] = 0.0f; acc1[r] = 0.0f; }

#pragma unroll
        for (int kc = 0; kc < 16; kc += 2) {
            // B 32x16 fragment, contiguous k order (same as A): lane L holds col L%16, K=(L/16)*16+{0..15}
            v16h b0 = *(const v16h*)(&buf[brow + kc * 32]);
            acc0 = __builtin_amdgcn_wmma_f32_16x16x32_f16(
                false, afrag[kc], false, b0, (short)0, acc0, false, false);
            v16h b1 = *(const v16h*)(&buf[brow + (kc + 1) * 32]);
            acc1 = __builtin_amdgcn_wmma_f32_16x16x32_f16(
                false, afrag[kc + 1], false, b1, (short)0, acc1, false, false);
        }

        // Epilogue: acc[r] on lane L = sim[hq = r + 8*(L/16)][hp = L%16]
        float ssum = 0.f;
#pragma unroll
        for (int r = 0; r < 8; ++r) {
            float v = acc0[r] + acc1[r];
            if (l16 >= NH) v = -INFINITY;            // mask padded hp cols
            v = fmaxf(v, __shfl_xor(v, 1, 32));
            v = fmaxf(v, __shfl_xor(v, 2, 32));
            v = fmaxf(v, __shfl_xor(v, 4, 32));
            v = fmaxf(v, __shfl_xor(v, 8, 32));      // max over 16-lane half
            int hqrow = r + half * 8;
            if (hqrow < NH) ssum += v;               // skip padded hq rows
        }
        ssum += __shfl_xor(ssum, 16, 32);            // low half + high half
        if (lane == 0) scores[(size_t)q * NCLASS + c] = ssum;

        // commit next tile to the other LDS buffer, then one barrier/class
        if (c + 1 < NCLASS)
            tile_lds_store(&ldsB[(c + 1) & 1][0], t, st);
        __syncthreads();
    }
}

// ---------------------------------------------------------------------------
// Kernel 4: in-place row log-softmax over 100 classes. One wave per row.
// ---------------------------------------------------------------------------
__global__ void lsm_kernel(float* __restrict__ inout) {
    int wave = threadIdx.x >> 5;
    int lane = threadIdx.x & 31;
    int row  = blockIdx.x * 8 + wave;
    float* s = inout + (size_t)row * NCLASS;

    float x[4];
    float mx = -INFINITY;
#pragma unroll
    for (int j = 0; j < 4; ++j) {
        int idx = lane + 32 * j;
        x[j] = (idx < NCLASS) ? s[idx] : -INFINITY;
        mx = fmaxf(mx, x[j]);
    }
    for (int m = 16; m > 0; m >>= 1) mx = fmaxf(mx, __shfl_xor(mx, m, 32));
    float se = 0.f;
#pragma unroll
    for (int j = 0; j < 4; ++j) {
        int idx = lane + 32 * j;
        se += (idx < NCLASS) ? expf(x[j] - mx) : 0.f;
    }
    for (int m = 16; m > 0; m >>= 1) se += __shfl_xor(se, m, 32);
    float lse = logf(se) + mx;
#pragma unroll
    for (int j = 0; j < 4; ++j) {
        int idx = lane + 32 * j;
        if (idx < NCLASS) s[idx] = x[j] - lse;
    }
}

// ---------------------------------------------------------------------------
extern "C" void kernel_launch(void* const* d_in, const int* in_sizes, int n_in,
                              void* d_out, int out_size, void* d_ws, size_t ws_size,
                              hipStream_t stream) {
    const float* zs = (const float*)d_in[0];
    const float* zq = (const float*)d_in[1];
    // d_in[2..6] are the shape scalars; fixed at compile time here.

    char* ws = (char*)d_ws;
    _Float16* protoH = (_Float16*)ws;                       // 1,638,400 B
    float*    invn   = (float*)(ws + (2 * 1024 * 1024));    //   256,000 B
    float*    out    = (float*)d_out;                       // scores -> logsoftmax in-place

    proto_kernel  <<<NCLASS * 16, 256, 0, stream>>>(zs, protoH);
    invnorm_kernel<<<ROWS / 8,    256, 0, stream>>>(zq, invn);
    gemm_kernel   <<<QTOT / 8,    256, 0, stream>>>(zq, invn, protoH, out);
    lsm_kernel    <<<QTOT / 8,    256, 0, stream>>>(out);
}
